// MinimalHPM_71803263255265
// MI455X (gfx1250) — matrix operand order repaired and checked by reading, not compiled, not run on hardware
//
#include <hip/hip_runtime.h>
#include <hip/hip_bf16.h>

// ---------------------------------------------------------------------------
// MinimalHPM.read on MI455X (gfx1250):
//   out[b,c] = sum_n exp(-2*r2(b,n) - 0.5*|t(b,n)|) * mem[n,c]
// Treated as a 16 x 2.1M x 16 GEMM with the A-matrix (Gaussian-tube kernel)
// generated on the fly and fed to V_WMMA_F32_16X16X32_F16; one 16x16 f32
// accumulator tile per wave, f32 accumulation throughout.
// HBM-bound: 128 MB of `memory` streamed once (~5.5 us at 23.3 TB/s).
// The exp uses the native base-2 transcendental with log2(e) folded into the
// FMA constants (no separate scale multiply).
// ---------------------------------------------------------------------------

typedef __attribute__((ext_vector_type(16))) _Float16 v16h;
typedef __attribute__((ext_vector_type(8)))  float    v8f;

#define DGRID 128
#define NPTS  (DGRID * DGRID * DGRID)   // 2,097,152 grid points
#define NCH   16                        // channels (WMMA N)
#define NRAYS 16                        // rays     (WMMA M)

// -2/ln(2) and -0.5/ln(2): fold k = exp(-2*r2 - 0.5*|t|) into exp2().
#define C_R2  (-2.8853900817779268f)
#define C_T   (-0.7213475204444817f)

__global__ __launch_bounds__(256)
void hpm_read_wmma(const float* __restrict__ mem,     // (NPTS, 16) f32
                   const float* __restrict__ ray_o,   // (16, 3)    f32
                   const float* __restrict__ ray_d,   // (16, 3)    f32
                   float* __restrict__ out)           // (16, 16)   f32, pre-zeroed
{
    const int lane  = threadIdx.x & 31;
    const int r     = lane & 15;   // ray index this lane serves (A rows / D rows)
    const int khalf = lane >> 4;   // which K-half of the wave this lane covers

    // --- per-ray constants (uniform per lane, read once; tiny) -------------
    const float ox = ray_o[3 * r + 0], oy = ray_o[3 * r + 1], oz = ray_o[3 * r + 2];
    const float dx = ray_d[3 * r + 0], dy = ray_d[3 * r + 1], dz = ray_d[3 * r + 2];
    const float od = ox * dx + oy * dy + oz * dz;
    const float o2 = ox * ox + oy * oy + oz * oz;
    const float d2 = dx * dx + dy * dy + dz * dz;
    const float e2 = 2.0f - d2;          // r2 = ||x-o||^2 - t^2*(2-d2)

    const int wavesPerBlock = blockDim.x >> 5;
    const int wave   = blockIdx.x * wavesPerBlock + (threadIdx.x >> 5);
    const int nwaves = gridDim.x * wavesPerBlock;

    v8f acc = {0.f, 0.f, 0.f, 0.f, 0.f, 0.f, 0.f, 0.f};  // 16x16 f32 C/D tile

    // ISA VGPR layout bases (CDNA5 ISA §7.12.2, wave32):
    //  A 16-bit 16x32: lane L holds K = {kh*8 + 0..7} and {16 + kh*8 + 0..7}
    //  B 16-bit 32x16: lane L holds K = {kh*8 + 0..7} and {16 + kh*8 + 0..7}, channel = L&15
    const int pbA = khalf * 8;
    const int kbB = khalf * 8;
    const int c   = lane & 15;

    for (int ch = wave; ch < (NPTS / 32); ch += nwaves) {
        const int base = ch << 5;                       // 32 consecutive points
        // grid[i,j,k] = (i,j,k); within a 32-aligned chunk i,j are constant
        const float fi  = (float)(base >> 14);
        const float fj  = (float)((base >> 7) & 127);
        const float fk0 = (float)(base & 127);          // z base (multiple of 32)

        // chunk-level per-ray constants; per-point work collapses to fmas
        const float t0  = fi * dx + fj * dy + fk0 * dz - od;                 // t(p=0)
        const float dx0 = fi * fi + fj * fj + fk0 * fk0
                        - 2.0f * (fi * ox + fj * oy + fk0 * oz) + o2;        // ||x-o||^2(p=0)
        const float pa  = 2.0f * (fk0 - oz);            // dx2(p) = dx0 + p*(p+pa)

        // ---- A tile: kernel weights; per eval ~6 VALU + 1 v_exp_f32 (trans
        // pipe co-executes with VALU, so exp is effectively free)
        v16h a;
        #pragma unroll
        for (int s = 0; s < 8; ++s) {
            {
                const float p   = (float)(pbA + s);
                const float t   = fmaf(p, dz, t0);
                const float dd  = fmaf(p, p + pa, dx0);    // ||x-o||^2
                const float rr  = fmaf(-e2, t * t, dd);    // radial r^2
                const float arg = fmaf(C_R2, rr, C_T * fabsf(t));
                a[s] = (_Float16)__builtin_amdgcn_exp2f(arg);
            }
            {
                const float p   = (float)(pbA + 16 + s);
                const float t   = fmaf(p, dz, t0);
                const float dd  = fmaf(p, p + pa, dx0);
                const float rr  = fmaf(-e2, t * t, dd);
                const float arg = fmaf(C_R2, rr, C_T * fabsf(t));
                a[8 + s] = (_Float16)__builtin_amdgcn_exp2f(arg);
            }
        }

        // ---- B tile: stream memory[n, c] (f32 -> f16). Each b32 load
        // instruction covers two full 64B lines across the two half-waves;
        // single-use stream -> non-temporal hint.
        const float* bp = mem + ((size_t)(base + kbB) * NCH + c);
        // prefetch this wave's next chunk (speculative; OOB silently dropped)
        __builtin_prefetch(bp + (size_t)nwaves * 32 * NCH, 0, 1);

        v16h b;
        #pragma unroll
        for (int e = 0; e < 16; ++e)
            b[e] = (_Float16)__builtin_nontemporal_load(bp + (e + (e >> 3) * 8) * NCH);

        // ---- D = A x B + C  (16x16x32, f32 accumulate); loop bound is
        // wave-uniform so EXEC is all-ones here as WMMA requires
        acc = __builtin_amdgcn_wmma_f32_16x16x32_f16(
                /*neg_a=*/false, a, /*neg_b=*/false, b,
                /*c_mod=*/(short)0, acc, /*reuse_a=*/false, /*reuse_b=*/false);
    }

    // C/D layout: VGPR v -> row (khalf*8 + v), col = lane&15
    const int row0 = khalf * 8;
    const int col  = lane & 15;
    #pragma unroll
    for (int v = 0; v < 8; ++v)
        atomicAdd(&out[(row0 + v) * NCH + col], acc[v]);
}

extern "C" void kernel_launch(void* const* d_in, const int* in_sizes, int n_in,
                              void* d_out, int out_size, void* d_ws, size_t ws_size,
                              hipStream_t stream) {
    (void)in_sizes; (void)n_in; (void)d_ws; (void)ws_size;
    const float* mem   = (const float*)d_in[0];   // memory   (128^3, 16) f32
    /* d_in[1] is the meshgrid — coordinates are recomputed from indices */
    const float* ray_o = (const float*)d_in[2];   // (16,3) f32
    const float* ray_d = (const float*)d_in[3];   // (16,3) f32
    float* out = (float*)d_out;                   // (16,16) f32

    // out is accumulated with atomics: zero it first (graph-capturable).
    hipMemsetAsync(out, 0, (size_t)out_size * sizeof(float), stream);

    // 65536 K-chunks of 32 points; 1024 blocks x 8 waves = 8192 waves
    // -> 8 chunks/wave, plenty of waves to saturate HBM across all WGPs.
    dim3 block(256);
    dim3 grid(1024);
    hpm_read_wmma<<<grid, block, 0, stream>>>(mem, ray_o, ray_d, out);
}
